// KalmanNetNN_74543452390111
// MI455X (gfx1250) — matrix operand order repaired and checked by reading, not compiled, not run on hardware
//
#include <hip/hip_runtime.h>

// ---------------------------------------------------------------------------
// KalmanNet persistent kernel for gfx1250 (MI455X).
// Batch elements are independent across the T=32 scan, so each workgroup owns
// a 16-row batch tile and iterates all timesteps, carrying (h, prior, post)
// in LDS. GRU / FC GEMMs run as v_wmma_f32_16x16x32_bf16 with weights
// pre-packed into B-fragment order in the workspace.
// ---------------------------------------------------------------------------

typedef __attribute__((ext_vector_type(16))) __bf16 v16bf;
typedef __attribute__((ext_vector_type(8)))  float  v8f;

#define T_STEPS 32
#define BATCH   8192
#define H1      480
#define HID     200
#define H2      320
#define BT      16      // batch rows per workgroup
#define KT1     15      // 480/32 k-tiles (a1 as A operand)
#define KTH     7       // ceil(200/32) k-tiles (h as A operand, K padded to 224)
#define NT_G    38      // ceil(600/16) n-tiles for GRU gate GEMMs
#define NT_2    20      // 320/16  n-tiles for a2 GEMM
#define GATE    600

// Workspace layout (bytes); packed bf16 B-fragments
#define WGP_OFF 0u
#define WGP_SZ  (NT_G * KT1 * 512u * 2u)   // 583,680
#define UGP_OFF (WGP_OFF + WGP_SZ)
#define UGP_SZ  (NT_G * KTH * 512u * 2u)   // 272,384
#define W2P_OFF (UGP_OFF + UGP_SZ)
#define W2P_SZ  (NT_2 * KTH * 512u * 2u)   // 143,360

// A-matrix 16x32 bf16 element index within a lane's v16bf, given k%32.
// ISA table: lanes 0-15 hold K {0..7,16..23}, lanes 16-31 hold K {8..15,24..31}.
__device__ __forceinline__ int frag_a_elem(int kin) {
  return (kin & 7) + ((kin & 16) ? 8 : 0);
}
__device__ __forceinline__ int frag_a_lane(int row, int kin) {
  return row + (((kin >> 3) & 1) << 4);
}

// ---------------------------------------------------------------------------
// Pack fp32 [Kdim x Ndim] row-major weight (i.e. B = KxN) into WMMA B-fragment
// order: [nt][kt][lane(32)][e(16)] bf16, zero padded. B layout (matches A):
// lanes 0-15 = N 0..15 with K = kt*32 + {0..7,16..23} (e), lanes 16-31 same N
// with K = kt*32 + {8..15,24..31}.
// ---------------------------------------------------------------------------
__global__ void pack_b_kernel(const float* __restrict__ W, __bf16* __restrict__ dst,
                              int Kdim, int Ndim, int KT, int total) {
  for (int i = blockIdx.x * blockDim.x + threadIdx.x; i < total;
       i += gridDim.x * blockDim.x) {
    int e    = i & 15;
    int lane = (i >> 4) & 31;
    int tile = i >> 9;
    int kt   = tile % KT;
    int nt   = tile / KT;
    int k = kt * 32 + ((e >> 3) << 4) + ((lane >> 4) << 3) + (e & 7);
    int c = nt * 16 + (lane & 15);
    float v = (k < Kdim && c < Ndim) ? W[k * Ndim + c] : 0.0f;
    dst[i] = (__bf16)v;
  }
}

// ---------------------------------------------------------------------------
// Persistent KalmanNet kernel: one workgroup = 16 batch rows through T steps.
// ---------------------------------------------------------------------------
__global__ __launch_bounds__(256, 1) void knet_kernel(
    const float* __restrict__ y,    // [T,B,2]
    const float* __restrict__ F,    // [4,4]
    const float* __restrict__ Hm,   // [2,4]
    const float* __restrict__ W1,   // [8,480]
    const float* __restrict__ b1,   // [480]
    const v16bf* __restrict__ WgP,  // packed bf16 fragments
    const v16bf* __restrict__ UgP,
    const float* __restrict__ bg,   // [2,600]
    const v16bf* __restrict__ W2P,
    const float* __restrict__ b2,   // [320]
    const float* __restrict__ W3,   // [320,8]
    const float* __restrict__ b3,   // [8]
    const float* __restrict__ hn0,  // [B,200]
    float* __restrict__ out)        // [T,B,4]
{
  __shared__ v16bf sAfrag[KT1][32];      // a1 tile, A-fragment order (bf16)
  __shared__ v16bf sHfrag[KTH][32];      // h  tile, A-fragment order (bf16)
  __shared__ float sXm[BT][NT_G * 16];   // a1 @ Wg   (fp32)
  __shared__ float sHm[BT][NT_G * 16];   // h  @ Ug   (fp32)
  __shared__ float sA2[BT][H2];
  __shared__ float sH[BT][HID];
  __shared__ float sKin[BT][8];
  __shared__ float sDy[BT][2];
  __shared__ float sPrior[BT][4];
  __shared__ float sPost[BT][4];
  __shared__ float sKG[BT][8];

  const int tid  = threadIdx.x;
  const int lane = tid & 31;
  const int wv   = tid >> 5;
  const int b0   = blockIdx.x * BT;

  // ---- init: zero padded h fragments, load hn0, zero states ----------------
  for (int i = tid; i < KTH * 32 * 16; i += 256)
    ((__bf16*)sHfrag)[i] = (__bf16)0.0f;
  for (int i = tid; i < BT * 4; i += 256) {
    (&sPrior[0][0])[i] = 0.0f;
    (&sPost[0][0])[i]  = 0.0f;
  }
  __syncthreads();
  for (int i = tid; i < BT * HID; i += 256) {
    int r = i / HID, k = i % HID;
    float v = hn0[(size_t)(b0 + r) * HID + k];
    sH[r][k] = v;
    ((__bf16*)&sHfrag[k >> 5][frag_a_lane(r, k & 31)])[frag_a_elem(k & 31)] = (__bf16)v;
  }
  __syncthreads();

  for (int t = 0; t < T_STEPS; ++t) {
    // ---- phase 1: state propagation + normalized features (16 lanes) ------
    if (tid < BT) {
      const int r = tid;
      float post[4], pprev[4], prior[4];
      #pragma unroll
      for (int m = 0; m < 4; ++m) { post[m] = sPost[r][m]; pprev[m] = sPrior[r][m]; }
      #pragma unroll
      for (int i = 0; i < 4; ++i) {
        float a = 0.0f;
        #pragma unroll
        for (int j = 0; j < 4; ++j) a += F[i * 4 + j] * post[j];
        prior[i] = a;
      }
      float m1y[2];
      #pragma unroll
      for (int i = 0; i < 2; ++i) {
        float a = 0.0f;
        #pragma unroll
        for (int j = 0; j < 4; ++j) a += Hm[i * 4 + j] * prior[j];
        m1y[i] = a;
      }
      float dx[4], ssx = 0.0f;
      #pragma unroll
      for (int m = 0; m < 4; ++m) { dx[m] = post[m] - pprev[m]; ssx += dx[m] * dx[m]; }
      float invx = rsqrtf(fmaxf(ssx, 1e-12f));
      float dy[2], ssy = 0.0f;
      #pragma unroll
      for (int i = 0; i < 2; ++i) {
        dy[i] = y[((size_t)t * BATCH + b0 + r) * 2 + i] - m1y[i];
        ssy += dy[i] * dy[i];
      }
      float invy = rsqrtf(fmaxf(ssy, 1e-12f));
      sKin[r][0] = dy[0] * invy; sKin[r][1] = dy[1] * invy;
      sKin[r][2] = dy[0] * invy; sKin[r][3] = dy[1] * invy;
      #pragma unroll
      for (int m = 0; m < 4; ++m) sKin[r][4 + m] = dx[m] * invx;
      sDy[r][0] = dy[0]; sDy[r][1] = dy[1];
      #pragma unroll
      for (int m = 0; m < 4; ++m) sPrior[r][m] = prior[m];
    }
    __syncthreads();

    // ---- phase 2: a1 = relu(knet_in @ W1 + b1), K=8, pack A-fragments ------
    for (int i = tid; i < BT * H1; i += 256) {
      int r = i / H1, c = i % H1;
      float acc = b1[c];
      #pragma unroll
      for (int k = 0; k < 8; ++k) acc += sKin[r][k] * W1[k * H1 + c];
      acc = fmaxf(acc, 0.0f);
      ((__bf16*)&sAfrag[c >> 5][frag_a_lane(r, c & 31)])[frag_a_elem(c & 31)] = (__bf16)acc;
    }
    __syncthreads();

    // ---- phase 3: xm = a1 @ Wg, hm = h @ Ug via WMMA bf16 ------------------
    for (int nt = wv; nt < NT_G; nt += 8) {
      v8f acc = {0.f, 0.f, 0.f, 0.f, 0.f, 0.f, 0.f, 0.f};
      #pragma unroll
      for (int kt = 0; kt < KT1; ++kt) {
        v16bf a  = sAfrag[kt][lane];
        v16bf bb = WgP[(nt * KT1 + kt) * 32 + lane];
        acc = __builtin_amdgcn_wmma_f32_16x16x32_bf16(false, a, false, bb,
                                                      (short)0, acc, false, false);
      }
      int row0 = (lane >> 4) << 3, col = nt * 16 + (lane & 15);
      #pragma unroll
      for (int v = 0; v < 8; ++v) sXm[row0 + v][col] = acc[v];
    }
    for (int nt = wv; nt < NT_G; nt += 8) {
      v8f acc = {0.f, 0.f, 0.f, 0.f, 0.f, 0.f, 0.f, 0.f};
      #pragma unroll
      for (int kt = 0; kt < KTH; ++kt) {
        v16bf a  = sHfrag[kt][lane];
        v16bf bb = UgP[(nt * KTH + kt) * 32 + lane];
        acc = __builtin_amdgcn_wmma_f32_16x16x32_bf16(false, a, false, bb,
                                                      (short)0, acc, false, false);
      }
      int row0 = (lane >> 4) << 3, col = nt * 16 + (lane & 15);
      #pragma unroll
      for (int v = 0; v < 8; ++v) sHm[row0 + v][col] = acc[v];
    }
    __syncthreads();

    // ---- phase 4: GRU gates (fp32), update h, repack h fragments -----------
    for (int i = tid; i < BT * HID; i += 256) {
      int r = i / HID, c = i % HID;
      float xz = sXm[r][c]            + bg[c];
      float xr = sXm[r][c + HID]      + bg[c + HID];
      float xh = sXm[r][c + 2 * HID]  + bg[c + 2 * HID];
      float hz = sHm[r][c]            + bg[GATE + c];
      float hr = sHm[r][c + HID]      + bg[GATE + c + HID];
      float hh = sHm[r][c + 2 * HID]  + bg[GATE + c + 2 * HID];
      float z  = 1.0f / (1.0f + __expf(-(xz + hz)));
      float rr = 1.0f / (1.0f + __expf(-(xr + hr)));
      float hc = tanhf(xh + rr * hh);
      float hn = z * sH[r][c] + (1.0f - z) * hc;
      sH[r][c] = hn;
      ((__bf16*)&sHfrag[c >> 5][frag_a_lane(r, c & 31)])[frag_a_elem(c & 31)] = (__bf16)hn;
    }
    __syncthreads();

    // ---- phase 5: a2 = relu(h @ W2 + b2) via WMMA --------------------------
    for (int nt = wv; nt < NT_2; nt += 8) {
      v8f acc = {0.f, 0.f, 0.f, 0.f, 0.f, 0.f, 0.f, 0.f};
      #pragma unroll
      for (int kt = 0; kt < KTH; ++kt) {
        v16bf a  = sHfrag[kt][lane];
        v16bf bb = W2P[(nt * KTH + kt) * 32 + lane];
        acc = __builtin_amdgcn_wmma_f32_16x16x32_bf16(false, a, false, bb,
                                                      (short)0, acc, false, false);
      }
      int row0 = (lane >> 4) << 3, col = nt * 16 + (lane & 15);
      #pragma unroll
      for (int v = 0; v < 8; ++v)
        sA2[row0 + v][col] = fmaxf(acc[v] + b2[col], 0.0f);
    }
    __syncthreads();

    // ---- phase 6: KG = a2 @ W3 + b3 (N=8, scalar) --------------------------
    if (tid < BT * 8) {
      int r = tid >> 3, j = tid & 7;
      float acc = b3[j];
      for (int k = 0; k < H2; ++k) acc += sA2[r][k] * W3[k * 8 + j];
      sKG[r][j] = acc;
    }
    __syncthreads();

    // ---- phase 7: innovation + posterior update + output -------------------
    if (tid < BT) {
      const int r = tid;
      float dy0 = sDy[r][0], dy1 = sDy[r][1];
      #pragma unroll
      for (int m = 0; m < 4; ++m) {
        float po = sPrior[r][m] + sKG[r][2 * m] * dy0 + sKG[r][2 * m + 1] * dy1;
        sPost[r][m] = po;
        out[((size_t)t * BATCH + b0 + r) * 4 + m] = po;
      }
    }
    __syncthreads();
  }
}

// ---------------------------------------------------------------------------
extern "C" void kernel_launch(void* const* d_in, const int* in_sizes, int n_in,
                              void* d_out, int out_size, void* d_ws, size_t ws_size,
                              hipStream_t stream) {
  const float* y   = (const float*)d_in[0];
  const float* F   = (const float*)d_in[1];
  const float* Hm  = (const float*)d_in[2];
  const float* W1  = (const float*)d_in[3];
  const float* b1  = (const float*)d_in[4];
  const float* Wg  = (const float*)d_in[5];
  const float* Ug  = (const float*)d_in[6];
  const float* bg  = (const float*)d_in[7];
  const float* W2  = (const float*)d_in[8];
  const float* b2  = (const float*)d_in[9];
  const float* W3  = (const float*)d_in[10];
  const float* b3  = (const float*)d_in[11];
  const float* hn0 = (const float*)d_in[12];
  float* out = (float*)d_out;

  __bf16* WgP = (__bf16*)((char*)d_ws + WGP_OFF);
  __bf16* UgP = (__bf16*)((char*)d_ws + UGP_OFF);
  __bf16* W2P = (__bf16*)((char*)d_ws + W2P_OFF);

  // Pack weights into WMMA B-fragment bf16 order (deterministic, every call).
  {
    int tot = NT_G * KT1 * 512;
    pack_b_kernel<<<(tot + 255) / 256, 256, 0, stream>>>(Wg, WgP, H1, GATE, KT1, tot);
  }
  {
    int tot = NT_G * KTH * 512;
    pack_b_kernel<<<(tot + 255) / 256, 256, 0, stream>>>(Ug, UgP, HID, GATE, KTH, tot);
  }
  {
    int tot = NT_2 * KTH * 512;
    pack_b_kernel<<<(tot + 255) / 256, 256, 0, stream>>>(W2, W2P, HID, H2, KTH, tot);
  }

  knet_kernel<<<BATCH / BT, 256, 0, stream>>>(
      y, F, Hm, W1, b1,
      (const v16bf*)WgP, (const v16bf*)UgP, bg,
      (const v16bf*)W2P, b2, W3, b3, hn0, out);
}
